// KSparseAutoencoder_33045478375540
// MI455X (gfx1250) — matrix operand order repaired and checked by reading, not compiled, not run on hardware
//
#include <hip/hip_runtime.h>
#include <float.h>

// ---------------------------------------------------------------------------
// KSAE forward for MI455X (gfx1250, wave32, WMMA).
//   B=8192 rows, V=2048 features, L=16384 latents, K(top)=32.
//   Encoder GEMM via v_wmma_f32_16x16x32_bf16 with fp32 hi/lo split (3 WMMAs
//   per K=32 step => ~fp32 product accuracy, fp32 accumulate).
//   Pre-split bf16 planes in workspace (conversion out of the hot loop),
//   B fragments loaded directly from global (no reuse => no LDS round-trip),
//   A tile double-buffered in LDS via GLOBAL_LOAD_ASYNC_TO_LDS.
// ---------------------------------------------------------------------------

typedef __attribute__((ext_vector_type(16))) __bf16        v16bf;
typedef __attribute__((ext_vector_type(8)))  float         v8f;
typedef __attribute__((ext_vector_type(4)))  float         f32x4;
typedef __attribute__((ext_vector_type(4)))  unsigned int  u32x4;
typedef __attribute__((ext_vector_type(2)))  unsigned int  u32x2;

union BFrag { v16bf v; u32x4 q[2]; };

__device__ __forceinline__ __bf16 bf_trunc(float f) {
  unsigned u = __builtin_bit_cast(unsigned, f);
  return __builtin_bit_cast(__bf16, (unsigned short)(u >> 16));
}
__device__ __forceinline__ float bf2f(__bf16 b) {
  unsigned u = ((unsigned)__builtin_bit_cast(unsigned short, b)) << 16;
  return __builtin_bit_cast(float, u);
}
__device__ __forceinline__ v8f wmma_bf(const BFrag& a, const BFrag& b, v8f c) {
  return __builtin_amdgcn_wmma_f32_16x16x32_bf16(false, a.v, false, b.v,
                                                 (short)0, c, false, false);
}

#define BATCH 8192
#define VEC   2048
#define LAT   16384
#define TOPK  32   // latentTopK (fixed by setup_inputs; == wave32 lane count)
#define KSTEPS (VEC / 32)

// ---- CDNA5 async global->LDS copy (guarded; sync fallback) ----------------
#if __has_builtin(__builtin_amdgcn_global_load_async_to_lds_b64)
#define HAVE_ASYNC_LDS 1
// hipcc's diagnostic told us the parameter type: vector_size(8) int pointers.
typedef int vs2i __attribute__((vector_size(8)));
typedef __attribute__((address_space(1))) vs2i* gbl_v2i_ptr;
typedef __attribute__((address_space(3))) vs2i* lds_v2i_ptr;
__device__ __forceinline__ void async_copy_b64(void* lds, const void* g) {
  __builtin_amdgcn_global_load_async_to_lds_b64(
      (gbl_v2i_ptr)g, (lds_v2i_ptr)lds, 0, 0);
}
#if __has_builtin(__builtin_amdgcn_s_wait_asynccnt)
#define WAIT_ASYNC(n) __builtin_amdgcn_s_wait_asynccnt(n)
#else
#define WAIT_ASYNC(n) asm volatile("s_wait_asynccnt %0" ::"n"(n) : "memory")
#endif
#endif

// ---------------------------------------------------------------------------
// Kernel 1: zero the f region of d_out (harness poisons it with 0xAA).
// ---------------------------------------------------------------------------
__global__ __launch_bounds__(256) void ksae_zero(f32x4* __restrict__ p, long long n4) {
  long long i = (long long)blockIdx.x * blockDim.x + threadIdx.x;
  const long long stride = (long long)gridDim.x * blockDim.x;
  f32x4 z = {0.f, 0.f, 0.f, 0.f};
  for (; i < n4; i += stride) p[i] = z;
}

// ---------------------------------------------------------------------------
// Kernel 1b (pre-pass): split fp32 -> bf16 (hi, lo) planes, optionally
// subtracting a broadcast row vector (xbar = x - dec_b). Runs once per call;
// removes all conversion VALU work from the 0.55 TFLOP hot loop.
// ---------------------------------------------------------------------------
__global__ __launch_bounds__(256) void ksae_split(
    const float* __restrict__ src, const float* __restrict__ sub,
    __bf16* __restrict__ hi, __bf16* __restrict__ lo, long long n)
{
  long long i = ((long long)blockIdx.x * blockDim.x + threadIdx.x) * 4;
  const long long stride = (long long)gridDim.x * blockDim.x * 4;
  for (; i < n; i += stride) {
    f32x4 v = *(const f32x4*)(src + i);
    if (sub) v -= *(const f32x4*)(sub + (i & (long long)(VEC - 1)));
#pragma unroll
    for (int e = 0; e < 4; ++e) {
      const __bf16 hb = bf_trunc(v[e]);
      hi[i + e] = hb;
      lo[i + e] = bf_trunc(v[e] - bf2f(hb));
    }
  }
}

// ---------------------------------------------------------------------------
// Kernel 2 (primary): encoder GEMM + streaming top-32, reading pre-split
// bf16 planes. Block = 256 threads (8 waves) owns 16 rows; loops over 128
// latent-column tiles; each wave computes one 16x16 WMMA tile of the slab.
//   - A (16x32 hi+lo, 2 KB) double-buffered in LDS via async copy
//   - B fragments: direct global_load_b128 from the weight planes (L2-hot)
//   - 3 x v_wmma_f32_16x16x32_bf16 per K=32 step, one barrier per step
// ---------------------------------------------------------------------------
__global__ __launch_bounds__(256) void ksae_encode_topk_pre(
    const __bf16* __restrict__ xhi, const __bf16* __restrict__ xlo,
    const __bf16* __restrict__ whi, const __bf16* __restrict__ wlo,
    const float* __restrict__ enc_b,
    float* __restrict__ wsval, int* __restrict__ wsidx)
{
  __shared__ __align__(16) __bf16 sA[2][2][16 * 32];  // [buf][hi/lo][r*32+k]
  __shared__ float sC[16 * 128];

  const int t    = threadIdx.x;
  const int lane = t & 31;
  const int w    = t >> 5;        // wave id 0..7 -> column sub-tile
  const int h    = lane >> 4;     // lane half (K-range select)
  const int n    = lane & 15;     // N index (B/C/D) == M index (A)
  const int R0   = blockIdx.x * 16;

  // A-staging assignment: each thread copies one 8-byte chunk.
  const int sp = t >> 7;                 // 0 = hi plane, 1 = lo plane
  const int so = (t & 127) * 8;          // byte offset in 1 KB plane tile
  const int sr = so >> 6;                // row 0..15
  const int sk = (so & 63) >> 1;         // k element 0..31 (multiple of 4)
  const __bf16* xsrc = sp ? xlo : xhi;

  auto stageA = [&](int k0, int buf) {
    const __bf16* g = xsrc + (size_t)(R0 + sr) * VEC + k0 + sk;
    void* l = &sA[buf][sp][sr * 32 + sk];
#ifdef HAVE_ASYNC_LDS
    async_copy_b64(l, g);
#else
    *(u32x2*)l = *(const u32x2*)g;
#endif
  };

  // running top-32 per row (wave w owns rows 2w and 2w+1); one slot per lane
  float bestv0 = -FLT_MAX, bestv1 = -FLT_MAX;
  int   besti0 = 0,        besti1 = 0;
  float thr0   = -FLT_MAX, thr1   = -FLT_MAX;

  for (int tile = 0; tile < LAT / 128; ++tile) {
    const int C0 = tile * 128;
    v8f acc = {};

    // prologue: stage A for ks = 0
    stageA(0, 0);
#ifdef HAVE_ASYNC_LDS
    WAIT_ASYNC(0);
#endif
    __syncthreads();

    for (int ks = 0; ks < KSTEPS; ++ks) {
      const int k0  = ks * 32;
      const int buf = ks & 1;
      if (ks + 1 < KSTEPS) stageA(k0 + 32, buf ^ 1);   // overlap with compute

      // ---- A fragments from LDS (contiguous k = 16*h + s, same order as B) ----
      BFrag Ah, Al;
      Ah.q[0] = *(const u32x4*)&sA[buf][0][n * 32 + h * 16];
      Ah.q[1] = *(const u32x4*)&sA[buf][0][n * 32 + h * 16 + 8];
      Al.q[0] = *(const u32x4*)&sA[buf][1][n * 32 + h * 16];
      Al.q[1] = *(const u32x4*)&sA[buf][1][n * 32 + h * 16 + 8];

      // ---- B fragments: direct from global bf16 planes (no reuse) ----
      const size_t be = (size_t)(C0 + (w << 4) + n) * VEC + k0 + h * 16;
      BFrag Bh, Bl;
      Bh.q[0] = *(const u32x4*)(whi + be);
      Bh.q[1] = *(const u32x4*)(whi + be + 8);
      Bl.q[0] = *(const u32x4*)(wlo + be);
      Bl.q[1] = *(const u32x4*)(wlo + be + 8);
      __builtin_prefetch(whi + be + 32, 0, 1);
      __builtin_prefetch(wlo + be + 32, 0, 1);

      // D = Ahi*Bhi + Ahi*Blo + Alo*Bhi + C
      acc = wmma_bf(Ah, Bh, acc);
      acc = wmma_bf(Ah, Bl, acc);
      acc = wmma_bf(Al, Bh, acc);

#ifdef HAVE_ASYNC_LDS
      if (ks + 1 < KSTEPS) WAIT_ASYNC(0);  // publish next A buffer
#endif
      __syncthreads();
    }

    // ---- bias add + spill 16x128 slab to LDS ----
    const float bias = enc_b[C0 + (w << 4) + n];
#pragma unroll
    for (int r = 0; r < 8; ++r)
      sC[(r + 8 * h) * 128 + (w << 4) + n] = acc[r] + bias;  // M = r + 8*half
    __syncthreads();

    // ---- streaming top-32 update: wave w handles rows 2w and 2w+1 ----
#pragma unroll
    for (int rr = 0; rr < 2; ++rr) {
      float bv = rr ? bestv1 : bestv0;
      int   bi = rr ? besti1 : besti0;
      float th = rr ? thr1   : thr0;
      const int row = (w << 1) | rr;

      float lm = -FLT_MAX;  // quick reject: wave-max of the 128 candidates
#pragma unroll
      for (int j = 0; j < 4; ++j)
        lm = fmaxf(lm, sC[row * 128 + lane * 4 + j]);
#pragma unroll
      for (int off = 16; off; off >>= 1)
        lm = fmaxf(lm, __shfl_xor(lm, off, 32));

      if (lm > th) {  // wave-uniform
        for (int cc = 0; cc < 128; ++cc) {
          const float cand = sC[row * 128 + cc];  // LDS broadcast, uniform
          if (cand > th) {
            const unsigned long long mk = __ballot(bv == th);
            const int src = __ffsll(mk) - 1;
            if (lane == src) { bv = cand; bi = C0 + cc; }
            float m2 = bv;
#pragma unroll
            for (int off = 16; off; off >>= 1)
              m2 = fminf(m2, __shfl_xor(m2, off, 32));
            th = m2;
          }
        }
      }
      if (rr) { bestv1 = bv; besti1 = bi; thr1 = th; }
      else    { bestv0 = bv; besti0 = bi; thr0 = th; }
    }
    __syncthreads();
  }

  const int row0 = R0 + (w << 1);
  wsval[(size_t)row0 * TOPK + lane]       = bestv0;
  wsidx[(size_t)row0 * TOPK + lane]       = besti0;
  wsval[(size_t)(row0 + 1) * TOPK + lane] = bestv1;
  wsidx[(size_t)(row0 + 1) * TOPK + lane] = besti1;
}

// ---------------------------------------------------------------------------
// Kernel 2 (fallback, small workspace): in-loop fp32->bf16 hi/lo conversion.
// Same WMMA core.
// ---------------------------------------------------------------------------
__global__ __launch_bounds__(256) void ksae_encode_topk(
    const float* __restrict__ x, const float* __restrict__ enc_w,
    const float* __restrict__ enc_b, const float* __restrict__ dec_b,
    float* __restrict__ wsval, int* __restrict__ wsidx)
{
  __shared__ __align__(16) __bf16 sAhi[16 * 32];
  __shared__ __align__(16) __bf16 sAlo[16 * 32];
  __shared__ __align__(16) __bf16 sBhi[128 * 32];
  __shared__ __align__(16) __bf16 sBlo[128 * 32];
  __shared__ float sC[16 * 128];

  const int t = threadIdx.x, lane = t & 31, w = t >> 5;
  const int h = lane >> 4, n = lane & 15;
  const int R0 = blockIdx.x * 16;

  float bestv0 = -FLT_MAX, bestv1 = -FLT_MAX;
  int   besti0 = 0,        besti1 = 0;
  float thr0   = -FLT_MAX, thr1   = -FLT_MAX;

  for (int tile = 0; tile < LAT / 128; ++tile) {
    const int C0 = tile * 128;
    v8f acc = {};

    for (int ks = 0; ks < KSTEPS; ++ks) {
      const int k0 = ks * 32;
      for (int i = t; i < 16 * 32; i += 256) {
        const int r = i >> 5, k = i & 31;
        const float f = x[(size_t)(R0 + r) * VEC + k0 + k] - dec_b[k0 + k];
        const __bf16 hi = bf_trunc(f);
        sAhi[i] = hi;
        sAlo[i] = bf_trunc(f - bf2f(hi));
      }
      {
        const int col = t >> 1, half = t & 1;
        const float* wr = enc_w + (size_t)(C0 + col) * VEC + k0 + half * 16;
        const int dst = col * 32 + half * 16;
#pragma unroll
        for (int q = 0; q < 4; ++q) {
          const f32x4 wv = *(const f32x4*)(wr + q * 4);
#pragma unroll
          for (int e = 0; e < 4; ++e) {
            const float f = wv[e];
            const __bf16 hi = bf_trunc(f);
            sBhi[dst + q * 4 + e] = hi;
            sBlo[dst + q * 4 + e] = bf_trunc(f - bf2f(hi));
          }
        }
        __builtin_prefetch(wr + 32, 0, 1);
      }
      __syncthreads();

      BFrag Ah, Al, Bh, Bl;
      Ah.q[0] = *(const u32x4*)&sAhi[n * 32 + h * 16];
      Ah.q[1] = *(const u32x4*)&sAhi[n * 32 + h * 16 + 8];
      Al.q[0] = *(const u32x4*)&sAlo[n * 32 + h * 16];
      Al.q[1] = *(const u32x4*)&sAlo[n * 32 + h * 16 + 8];
      const int c = (w << 4) + n;
      Bh.q[0] = *(const u32x4*)&sBhi[c * 32 + h * 16];
      Bh.q[1] = *(const u32x4*)&sBhi[c * 32 + h * 16 + 8];
      Bl.q[0] = *(const u32x4*)&sBlo[c * 32 + h * 16];
      Bl.q[1] = *(const u32x4*)&sBlo[c * 32 + h * 16 + 8];

      acc = wmma_bf(Ah, Bh, acc);
      acc = wmma_bf(Ah, Bl, acc);
      acc = wmma_bf(Al, Bh, acc);
      __syncthreads();
    }

    const float bias = enc_b[C0 + (w << 4) + n];
#pragma unroll
    for (int r = 0; r < 8; ++r)
      sC[(r + 8 * h) * 128 + (w << 4) + n] = acc[r] + bias;
    __syncthreads();

#pragma unroll
    for (int rr = 0; rr < 2; ++rr) {
      float bv = rr ? bestv1 : bestv0;
      int   bi = rr ? besti1 : besti0;
      float th = rr ? thr1   : thr0;
      const int row = (w << 1) | rr;

      float lm = -FLT_MAX;
#pragma unroll
      for (int j = 0; j < 4; ++j)
        lm = fmaxf(lm, sC[row * 128 + lane * 4 + j]);
#pragma unroll
      for (int off = 16; off; off >>= 1)
        lm = fmaxf(lm, __shfl_xor(lm, off, 32));

      if (lm > th) {
        for (int cc = 0; cc < 128; ++cc) {
          const float cand = sC[row * 128 + cc];
          if (cand > th) {
            const unsigned long long mk = __ballot(bv == th);
            const int src = __ffsll(mk) - 1;
            if (lane == src) { bv = cand; bi = C0 + cc; }
            float m2 = bv;
#pragma unroll
            for (int off = 16; off; off >>= 1)
              m2 = fminf(m2, __shfl_xor(m2, off, 32));
            th = m2;
          }
        }
      }
      if (rr) { bestv1 = bv; besti1 = bi; thr1 = th; }
      else    { bestv0 = bv; besti0 = bi; thr0 = th; }
    }
    __syncthreads();
  }

  const int row0 = R0 + (w << 1);
  wsval[(size_t)row0 * TOPK + lane]       = bestv0;
  wsidx[(size_t)row0 * TOPK + lane]       = besti0;
  wsval[(size_t)(row0 + 1) * TOPK + lane] = bestv1;
  wsidx[(size_t)(row0 + 1) * TOPK + lane] = besti1;
}

// ---------------------------------------------------------------------------
// Kernel 3: scatter relu(top-k) into f; sparse decode xhat = f @ dec_w.T + b.
// One block per row; dec_w (128 MB) stays resident in the 192 MB L2.
// ---------------------------------------------------------------------------
__global__ __launch_bounds__(256) void ksae_scatter_decode(
    const float* __restrict__ dec_w, const float* __restrict__ dec_b,
    const float* __restrict__ wsval, const int* __restrict__ wsidx,
    float* __restrict__ f, float* __restrict__ xhat)
{
  __shared__ float sv[TOPK];
  __shared__ int   si[TOPK];
  const int b = blockIdx.x;
  const int t = threadIdx.x;

  if (t < TOPK) {
    const float v  = wsval[(size_t)b * TOPK + t];
    const int   ix = wsidx[(size_t)b * TOPK + t];
    const float rv = fmaxf(v, 0.0f);
    sv[t] = rv;
    si[t] = ix;
    f[(size_t)b * LAT + ix] = rv;   // scatter into (pre-zeroed) f
  }
  __syncthreads();

  for (int v = t; v < VEC; v += 256) {
    float acc = dec_b[v];
    const float* wr = dec_w + (size_t)v * LAT;
#pragma unroll
    for (int j = 0; j < TOPK; ++j)
      acc += sv[j] * wr[si[j]];
    xhat[(size_t)b * VEC + v] = acc;
  }
}

// ---------------------------------------------------------------------------
extern "C" void kernel_launch(void* const* d_in, const int* in_sizes, int n_in,
                              void* d_out, int out_size, void* d_ws, size_t ws_size,
                              hipStream_t stream) {
  (void)in_sizes; (void)n_in; (void)out_size;

  const float* x     = (const float*)d_in[0];
  const float* enc_w = (const float*)d_in[1];
  const float* enc_b = (const float*)d_in[2];
  const float* dec_w = (const float*)d_in[3];
  const float* dec_b = (const float*)d_in[4];
  // d_in[5] = latentTopK (== 32, fixed by the reference setup)

  float* f    = (float*)d_out;                       // [8192, 16384]
  float* xhat = f + (size_t)BATCH * LAT;             // [8192, 2048]

  float* wsval = (float*)d_ws;                                   // 1 MB
  int*   wsidx = (int*)((char*)d_ws + (size_t)BATCH * TOPK * 4); // 1 MB

  const long long n4 = (long long)BATCH * LAT / 4;   // f region in float4s
  ksae_zero<<<2048, 256, 0, stream>>>((f32x4*)d_out, n4);

  const size_t planes_off = (size_t)2 * BATCH * TOPK * 4;        // 2 MB
  const size_t need = planes_off +
                      (size_t)LAT * VEC * 2 * 2 +                // enc_w hi/lo
                      (size_t)BATCH * VEC * 2 * 2;               // xbar hi/lo
  if (ws_size >= need) {
    __bf16* whi = (__bf16*)((char*)d_ws + planes_off);
    __bf16* wlo = whi + (size_t)LAT * VEC;
    __bf16* xhi = wlo + (size_t)LAT * VEC;
    __bf16* xlo = xhi + (size_t)BATCH * VEC;
    ksae_split<<<4096, 256, 0, stream>>>(enc_w, nullptr, whi, wlo,
                                         (long long)LAT * VEC);
    ksae_split<<<2048, 256, 0, stream>>>(x, dec_b, xhi, xlo,
                                         (long long)BATCH * VEC);
    ksae_encode_topk_pre<<<BATCH / 16, 256, 0, stream>>>(xhi, xlo, whi, wlo,
                                                         enc_b, wsval, wsidx);
  } else {
    ksae_encode_topk<<<BATCH / 16, 256, 0, stream>>>(x, enc_w, enc_b, dec_b,
                                                     wsval, wsidx);
  }

  ksae_scatter_decode<<<BATCH, 256, 0, stream>>>(dec_w, dec_b, wsval, wsidx,
                                                 f, xhat);
}
